// AttnPooling_61057255080016
// MI455X (gfx1250) — matrix operand order repaired and checked by reading, not compiled, not run on hardware
//
#include <hip/hip_runtime.h>
#include <cstdint>
#include <cstddef>

// Problem constants (match reference)
#define BB   2
#define NN   2047
#define LL   2048   // N+1 (pool token prepended)
#define DD   1024
#define HH   16
#define HD   64
#define NLAY 2
#define DFF  4096

typedef __attribute__((ext_vector_type(16))) __bf16    v16bf;
typedef __attribute__((ext_vector_type(8)))  float     v8f;
typedef __attribute__((ext_vector_type(4)))  uint32_t  v4u;
typedef __attribute__((ext_vector_type(8)))  uint32_t  v8u;

__device__ __forceinline__ v8f wmma_bf16(v16bf a, v16bf b, v8f c) {
  return __builtin_amdgcn_wmma_f32_16x16x32_bf16(false, a, false, b, (short)0, c,
                                                 false, false);
}

// A-fragment: 16x32 bf16 tile (row-major, ld = lda). Lane l holds row
// row0+(l&15); lanes 0-15 hold K k0+0..15, lanes 16-31 hold
// K k0+16..31 (element i is K = k0+(l>>4)*16+i). Two b128 loads.
__device__ __forceinline__ v16bf load_a16x32(const __bf16* __restrict__ A, int lda,
                                             int row0, int k0, int lane, int rowMax) {
  int m = row0 + (lane & 15);
  if (m > rowMax) m = rowMax;
  const __bf16* p = A + (size_t)m * lda + k0 + ((lane >> 4) << 4);
  union { v16bf v; uint4 u[2]; } t;
  t.u[0] = *(const uint4*)p;
  t.u[1] = *(const uint4*)(p + 8);
  return t.v;
}

// B-fragment (32x16) for C = A * S^T where S is row-major (n-major, ld = lds):
// B[k][n] = S[n][k]. Lane l holds col n0+(l&15); element i is K = k0+(l>>4)*16+i.
// Works for global weights and LDS tiles alike. Two b128 loads.
__device__ __forceinline__ v16bf load_b32x16(const __bf16* __restrict__ S, int lds,
                                             int n0, int k0, int lane) {
  const __bf16* p = S + (size_t)(n0 + (lane & 15)) * lds + k0 + ((lane >> 4) << 4);
  union { v16bf v; uint4 u[2]; } t;
  t.u[0] = *(const uint4*)p;
  t.u[1] = *(const uint4*)(p + 8);
  return t.v;
}

// ------------------------------------------------------------- TDM tile load
// Issue a Tensor Data Mover 2D tile load (rows x rowlen bf16, global row
// stride strideElems) into LDS at byte offset ldsOff. D# per ISA §8:
// group0 = {count=1, lds_addr, global_addr, type=2}; group1 = {mask=0,
// data_size=1(2B), tensor_dim0=rowlen, tensor_dim1=rows, tile_dim0=rowlen,
// tile_dim1=rows, dim0_stride=strideElems}. Groups 2/3 zero (2D tile).
// Tracked by TENSORcnt.
__device__ __forceinline__ void tdm_load_2d(const void* gsrc, uint32_t ldsOff,
                                            uint32_t rows, uint32_t rowlen,
                                            uint32_t strideElems) {
  uint64_t ga = (uint64_t)(uintptr_t)gsrc;
  v4u g0;
  g0[0] = 1u;                                               // count=1 (valid user D#)
  g0[1] = ldsOff;                                           // lds_addr (bytes)
  g0[2] = (uint32_t)ga;                                     // global_addr[31:0]
  g0[3] = ((uint32_t)(ga >> 32) & 0x01FFFFFFu) | (2u << 30);// addr[56:32] | type=2
  v8u g1;
  g1[0] = 1u << 16;                                         // data_size=1 -> 2 bytes
  g1[1] = (rowlen & 0xFFFFu) << 16;                         // tensor_dim0[15:0]
  g1[2] = (rowlen >> 16) | ((rows & 0xFFFFu) << 16);        // dim0[31:16] | dim1[15:0]
  g1[3] = (rows >> 16) | ((rowlen & 0xFFFFu) << 16);        // dim1[31:16] | tile_dim0
  g1[4] = rows & 0xFFFFu;                                   // tile_dim1 (tile_dim2=0)
  g1[5] = strideElems;                                      // tensor_dim0_stride[31:0]
  g1[6] = 0u;
  g1[7] = 0u;
  v4u gz = {0u, 0u, 0u, 0u};
  asm volatile("tensor_load_to_lds %0, %1, %2, %3"
               :: "s"(g0), "s"(g1), "s"(gz), "s"(gz)
               : "memory");
}

// ---------------------------------------------------------------- prep kernels
__global__ void prep_xc(const float* __restrict__ x, const float* __restrict__ pool,
                        float* __restrict__ xc) {
  size_t idx = (size_t)blockIdx.x * blockDim.x + threadIdx.x;  // over B*L*D
  size_t b = idx / ((size_t)LL * DD);
  size_t r = idx % ((size_t)LL * DD);
  size_t l = r / DD, d = r % DD;
  xc[idx] = (l == 0) ? pool[d] : x[((size_t)b * NN + (l - 1)) * DD + d];
}

__global__ void prep_mask(const int* __restrict__ am, float* __restrict__ maskadd) {
  int idx = blockIdx.x * blockDim.x + threadIdx.x;  // over B*L
  if (idx >= BB * LL) return;
  int b = idx / LL, l = idx % LL;
  float v = 0.0f;
  if (l > 0 && am[b * NN + (l - 1)] == 0) v = -1.0e30f;
  maskadd[idx] = v;
}

__global__ void cvt_bf16(const float* __restrict__ in, __bf16* __restrict__ out, size_t n) {
  size_t stride = (size_t)gridDim.x * blockDim.x;
  for (size_t i = (size_t)blockIdx.x * blockDim.x + threadIdx.x; i < n; i += stride)
    out[i] = (__bf16)in[i];
}

// ---------------------------------------------------------------- layernorm
__global__ void layernorm_bf16(const float* __restrict__ x, size_t rstride,
                               const float* __restrict__ g, const float* __restrict__ bta,
                               __bf16* __restrict__ out, int D) {
  __shared__ float s1[256], s2[256];
  int row = blockIdx.x, tid = threadIdx.x;
  const float* xr = x + (size_t)row * rstride;
  float a = 0.f, q = 0.f;
  for (int i = tid; i < D; i += 256) { float v = xr[i]; a += v; q += v * v; }
  s1[tid] = a; s2[tid] = q;
  __syncthreads();
  for (int s = 128; s > 0; s >>= 1) {
    if (tid < s) { s1[tid] += s1[tid + s]; s2[tid] += s2[tid + s]; }
    __syncthreads();
  }
  float mean = s1[0] / D;
  float var  = s2[0] / D - mean * mean;
  float inv  = rsqrtf(var + 1e-5f);
  __bf16* orow = out + (size_t)row * D;
  for (int i = tid; i < D; i += 256)
    orow[i] = (__bf16)((xr[i] - mean) * inv * g[i] + bta[i]);
}

// ---------------------------------------------------------------- GEMM: C = A * W^T
// A: (M,K) bf16 row-major, W: (N,K) bf16 row-major. Block = 128 thr (4 waves).
// Each wave owns a 32x64 register tile: 2 A-frags x 4 B-frags -> 8 WMMA per
// K-step over 6 fragment loads. Epilogue fuses bias / exact GELU / residual.
template <bool HAS_BIAS, bool HAS_RES, bool DO_GELU, bool OUT_BF16>
__global__ void __launch_bounds__(128)
gemm_bf16(const __bf16* __restrict__ A, const __bf16* __restrict__ W,
          const float* __restrict__ bias, const float* __restrict__ res,
          void* __restrict__ outp, int M, int N, int K) {
  int lane = threadIdx.x & 31;
  int wave = threadIdx.x >> 5;
  int n0 = blockIdx.x * 256 + wave * 64;
  int m0 = blockIdx.y * 32;
  v8f c0[4] = {}, c1[4] = {};
  const __bf16* pa = A + (size_t)min(m0 + (lane & 15), M - 1) * K;
  const __bf16* pw = W + (size_t)(n0 + (lane & 15)) * K;
  for (int k0 = 0; k0 < K; k0 += 32) {
    if (k0 + 32 < K) {                    // next K-slice -> global_prefetch_b8
      __builtin_prefetch(pa + k0 + 32, 0, 1);
      __builtin_prefetch(pw + k0 + 32, 0, 1);
    }
    v16bf a0 = load_a16x32(A, K, m0,      k0, lane, M - 1);
    v16bf a1 = load_a16x32(A, K, m0 + 16, k0, lane, M - 1);
#pragma unroll
    for (int j = 0; j < 4; ++j) {
      v16bf b = load_b32x16(W, K, n0 + j * 16, k0, lane);
      c0[j] = wmma_bf16(a0, b, c0[j]);
      c1[j] = wmma_bf16(a1, b, c1[j]);
    }
  }
  int nl = lane & 15, hf = lane >> 4;
  float* outF = (float*)outp;
  __bf16* outB = (__bf16*)outp;
#pragma unroll
  for (int j = 0; j < 4; ++j) {
    int n = n0 + j * 16 + nl;
    float bv = HAS_BIAS ? bias[n] : 0.f;
#pragma unroll
    for (int r = 0; r < 8; ++r) {
#pragma unroll
      for (int half = 0; half < 2; ++half) {
        int m = m0 + half * 16 + r + hf * 8;
        if (m >= M) continue;
        float v = (half ? c1[j][r] : c0[j][r]) + bv;
        if (DO_GELU) v = 0.5f * v * (1.f + erff(v * 0.70710678118654752f));
        if (HAS_RES) v += res[(size_t)m * N + n];
        if (OUT_BF16) outB[(size_t)m * N + n] = (__bf16)v;
        else          outF[(size_t)m * N + n] = v;
      }
    }
  }
}

// ---------------------------------------------------------------- RoPE (GPT-NeoX)
__global__ void rope_qk(__bf16* __restrict__ qb, __bf16* __restrict__ kb) {
  size_t total = (size_t)BB * (LL - 1) * HH * (HD / 2);
  size_t idx = (size_t)blockIdx.x * blockDim.x + threadIdx.x;
  if (idx >= total) return;
  int i = idx % (HD / 2); idx /= (HD / 2);
  int h = idx % HH;       idx /= HH;
  int l = 1 + (int)(idx % (LL - 1));
  int b = (int)(idx / (LL - 1));
  float inv = __expf(-logf(10000.f) * (float)i / (float)(HD / 2));
  float fr = (float)l * inv;  // offset=1 -> angle position == token index l
  float cs = cosf(fr), sn = sinf(fr);
  size_t base = ((size_t)b * LL + l) * DD + (size_t)h * HD;
  float q1 = (float)qb[base + i], q2 = (float)qb[base + i + HD / 2];
  qb[base + i]          = (__bf16)(q1 * cs - q2 * sn);
  qb[base + i + HD / 2] = (__bf16)(q2 * cs + q1 * sn);
  float k1 = (float)kb[base + i], k2 = (float)kb[base + i + HD / 2];
  kb[base + i]          = (__bf16)(k1 * cs - k2 * sn);
  kb[base + i + HD / 2] = (__bf16)(k2 * cs + k1 * sn);
}

// ---------------------------------------------------------------- flash attention
// One wave per (b, h, 16-query tile). K/V 32x64 tiles are staged into LDS by
// the Tensor Data Mover (double-buffered, TENSORcnt-synchronized); S = Q*K^T
// and P*V run on WMMA; online softmax uses 16-lane shfl reductions; P is
// re-laid-out through a 1KB LDS tile into an A-fragment.
__global__ void __launch_bounds__(32)
flash_attn(const __bf16* __restrict__ Q, const __bf16* __restrict__ K,
           const __bf16* __restrict__ V, const float* __restrict__ maskadd,
           __bf16* __restrict__ O) {
  __shared__ __align__(16) __bf16 kbuf[2][32 * 64];
  __shared__ __align__(16) __bf16 vbuf[2][32 * 64];
  __shared__ __align__(16) __bf16 plds[16 * 32];
  int lane = threadIdx.x;
  int qt = blockIdx.x, h = blockIdx.y, b = blockIdx.z;
  const __bf16* Qb = Q + ((size_t)b * LL) * DD + (size_t)h * HD;
  const __bf16* Kb = K + ((size_t)b * LL) * DD + (size_t)h * HD;
  const __bf16* Vb = V + ((size_t)b * LL) * DD + (size_t)h * HD;
  const float*  mk = maskadd + (size_t)b * LL;

  int m0  = qt * 16;
  int hf  = lane >> 4;
  int nlo = lane & 15;

  v16bf aq0 = load_a16x32(Qb, DD, m0, 0,  lane, LL - 1);
  v16bf aq1 = load_a16x32(Qb, DD, m0, 32, lane, LL - 1);

  float rm[8], rl[8];
#pragma unroll
  for (int r = 0; r < 8; ++r) { rm[r] = -1.0e30f; rl[r] = 0.f; }
  v8f o[4] = {};

  uint32_t kOff[2] = { (uint32_t)(uintptr_t)&kbuf[0][0], (uint32_t)(uintptr_t)&kbuf[1][0] };
  uint32_t vOff[2] = { (uint32_t)(uintptr_t)&vbuf[0][0], (uint32_t)(uintptr_t)&vbuf[1][0] };

  const int NIT = LL / 32;
  tdm_load_2d(Kb, kOff[0], 32, 64, DD);
  tdm_load_2d(Vb, vOff[0], 32, 64, DD);

  for (int it = 0; it < NIT; ++it) {
    int kb0 = it * 32;
    int cur = it & 1;
    if (it + 1 < NIT) {                       // prefetch next tiles via TDM
      tdm_load_2d(Kb + (size_t)(kb0 + 32) * DD, kOff[cur ^ 1], 32, 64, DD);
      tdm_load_2d(Vb + (size_t)(kb0 + 32) * DD, vOff[cur ^ 1], 32, 64, DD);
      __builtin_amdgcn_s_wait_tensorcnt(2);   // current pair landed
    } else {
      __builtin_amdgcn_s_wait_tensorcnt(0);
    }
    asm volatile("" ::: "memory");            // keep LDS reads below the wait
    const __bf16* kt = &kbuf[cur][0];
    const __bf16* vt = &vbuf[cur][0];

    v8f s0 = {}, s1 = {};
    s0 = wmma_bf16(aq0, load_b32x16(kt, 64, 0,  0,  lane), s0);
    s0 = wmma_bf16(aq1, load_b32x16(kt, 64, 0,  32, lane), s0);
    s1 = wmma_bf16(aq0, load_b32x16(kt, 64, 16, 0,  lane), s1);
    s1 = wmma_bf16(aq1, load_b32x16(kt, 64, 16, 32, lane), s1);
    float mk0 = mk[kb0 + nlo];
    float mk1 = mk[kb0 + 16 + nlo];
#pragma unroll
    for (int r = 0; r < 8; ++r) {
      float x0 = s0[r] * 0.125f + mk0;  // 1/sqrt(64)
      float x1 = s1[r] * 0.125f + mk1;
      float mx = fmaxf(x0, x1);
      mx = fmaxf(mx, __shfl_xor(mx, 1));  // rows live in 16-lane halves:
      mx = fmaxf(mx, __shfl_xor(mx, 2));  // reduce only within the half
      mx = fmaxf(mx, __shfl_xor(mx, 4));
      mx = fmaxf(mx, __shfl_xor(mx, 8));
      float mnew = fmaxf(rm[r], mx);
      float alp  = __expf(rm[r] - mnew);
      rm[r] = mnew;
      float p0 = __expf(x0 - mnew);
      float p1 = __expf(x1 - mnew);
      float ts = p0 + p1;
      ts += __shfl_xor(ts, 1); ts += __shfl_xor(ts, 2);
      ts += __shfl_xor(ts, 4); ts += __shfl_xor(ts, 8);
      rl[r] = rl[r] * alp + ts;
#pragma unroll
      for (int j = 0; j < 4; ++j) o[j][r] *= alp;
      int mrow = r + hf * 8;
      plds[mrow * 32 + nlo]      = (__bf16)p0;   // C-layout -> LDS tile
      plds[mrow * 32 + 16 + nlo] = (__bf16)p1;
    }
    asm volatile("s_wait_dscnt 0" ::: "memory");
    // Re-read P in A-fragment layout (row = lane&15).
    v16bf pa = load_a16x32(plds, 32, 0, 0, lane, 15);
    // P (16x32) * V (32x64): B-frag from LDS V tile (col n = out dim).
#pragma unroll
    for (int dt = 0; dt < 4; ++dt) {
      const __bf16* vp = vt + (size_t)(hf * 16) * 64 + dt * 16 + nlo;
      v16bf bv;
#pragma unroll
      for (int i = 0; i < 16; ++i) bv[i] = vp[(size_t)i * 64];
      o[dt] = wmma_bf16(pa, bv, o[dt]);
    }
  }
#pragma unroll
  for (int r = 0; r < 8; ++r) {
    float invl = 1.f / rl[r];
    int m = m0 + r + hf * 8;
    __bf16* orow = O + ((size_t)b * LL + m) * DD + (size_t)h * HD;
#pragma unroll
    for (int dt = 0; dt < 4; ++dt)
      orow[dt * 16 + nlo] = (__bf16)(o[dt][r] * invl);
  }
}

// ---------------------------------------------------------------- host driver
extern "C" void kernel_launch(void* const* d_in, const int* in_sizes, int n_in,
                              void* d_out, int out_size, void* d_ws, size_t ws_size,
                              hipStream_t stream) {
  (void)in_sizes; (void)n_in; (void)out_size; (void)ws_size;
  const float* x     = (const float*)d_in[0];
  const int*   am    = (const int*)  d_in[1];
  const float* pool  = (const float*)d_in[2];
  const float* Wq    = (const float*)d_in[3];
  const float* Wk    = (const float*)d_in[4];
  const float* Wv    = (const float*)d_in[5];
  const float* Wo    = (const float*)d_in[6];
  const float* ln1g  = (const float*)d_in[7];
  const float* ln1b  = (const float*)d_in[8];
  const float* ln2g  = (const float*)d_in[9];
  const float* ln2b  = (const float*)d_in[10];
  const float* W1    = (const float*)d_in[11];
  const float* b1    = (const float*)d_in[12];
  const float* W2    = (const float*)d_in[13];
  const float* b2    = (const float*)d_in[14];
  const float* outg  = (const float*)d_in[15];
  const float* outb  = (const float*)d_in[16];
  const float* projW = (const float*)d_in[17];
  const float* projb = (const float*)d_in[18];

  char* wsp = (char*)d_ws;
  auto carve = [&](size_t bytes) -> void* {
    void* p = (void*)wsp;
    wsp += (bytes + 255) & ~(size_t)255;
    return p;
  };
  const size_t BLD = (size_t)BB * LL * DD;           // 4096*1024
  float*  xc    = (float*) carve(BLD * 4);
  __bf16* xnb   = (__bf16*)carve(BLD * 2);
  __bf16* qb    = (__bf16*)carve(BLD * 2);
  __bf16* kb    = (__bf16*)carve(BLD * 2);
  __bf16* vb    = (__bf16*)carve(BLD * 2);
  __bf16* ob    = (__bf16*)carve(BLD * 2);
  __bf16* hmid  = (__bf16*)carve((size_t)BB * LL * DFF * 2);
  float*  maskA = (float*) carve((size_t)BB * LL * 4);
  __bf16* pooled= (__bf16*)carve((size_t)BB * DD * 2);
  __bf16* wqb   = (__bf16*)carve((size_t)NLAY * DD * DD * 2);
  __bf16* wkb   = (__bf16*)carve((size_t)NLAY * DD * DD * 2);
  __bf16* wvb   = (__bf16*)carve((size_t)NLAY * DD * DD * 2);
  __bf16* wob   = (__bf16*)carve((size_t)NLAY * DD * DD * 2);
  __bf16* w1b   = (__bf16*)carve((size_t)NLAY * DFF * DD * 2);
  __bf16* w2b   = (__bf16*)carve((size_t)NLAY * DD * DFF * 2);
  __bf16* pwb   = (__bf16*)carve((size_t)DD * DD * 2);

  // Inputs -> xc, mask, bf16 weights
  prep_xc<<<(unsigned)(BLD / 256), 256, 0, stream>>>(x, pool, xc);
  prep_mask<<<(BB * LL + 255) / 256, 256, 0, stream>>>(am, maskA);
  cvt_bf16<<<1024, 256, 0, stream>>>(Wq, wqb, (size_t)NLAY * DD * DD);
  cvt_bf16<<<1024, 256, 0, stream>>>(Wk, wkb, (size_t)NLAY * DD * DD);
  cvt_bf16<<<1024, 256, 0, stream>>>(Wv, wvb, (size_t)NLAY * DD * DD);
  cvt_bf16<<<1024, 256, 0, stream>>>(Wo, wob, (size_t)NLAY * DD * DD);
  cvt_bf16<<<1024, 256, 0, stream>>>(W1, w1b, (size_t)NLAY * DFF * DD);
  cvt_bf16<<<1024, 256, 0, stream>>>(W2, w2b, (size_t)NLAY * DD * DFF);
  cvt_bf16<<<1024, 256, 0, stream>>>(projW, pwb, (size_t)DD * DD);

  const int M = BB * LL;                 // 4096 token rows
  dim3 gD(DD / 256, M / 32);             // N=1024 GEMMs (block tile 32x256)
  dim3 gF(DFF / 256, M / 32);            // N=4096 GEMM
  dim3 gp(DD / 256, 1);                  // final projection (M=2, row guards)
  size_t ropeTotal = (size_t)BB * (LL - 1) * HH * (HD / 2);

  for (int i = 0; i < NLAY; ++i) {
    const __bf16* wq_i = wqb + (size_t)i * DD * DD;
    const __bf16* wk_i = wkb + (size_t)i * DD * DD;
    const __bf16* wv_i = wvb + (size_t)i * DD * DD;
    const __bf16* wo_i = wob + (size_t)i * DD * DD;
    const __bf16* w1_i = w1b + (size_t)i * DFF * DD;
    const __bf16* w2_i = w2b + (size_t)i * DD * DFF;

    layernorm_bf16<<<M, 256, 0, stream>>>(xc, (size_t)DD, ln1g + (size_t)i * DD,
                                          ln1b + (size_t)i * DD, xnb, DD);
    gemm_bf16<false,false,false,true><<<gD, 128, 0, stream>>>(xnb, wq_i, nullptr, nullptr, qb, M, DD, DD);
    gemm_bf16<false,false,false,true><<<gD, 128, 0, stream>>>(xnb, wk_i, nullptr, nullptr, kb, M, DD, DD);
    gemm_bf16<false,false,false,true><<<gD, 128, 0, stream>>>(xnb, wv_i, nullptr, nullptr, vb, M, DD, DD);
    rope_qk<<<(unsigned)((ropeTotal + 255) / 256), 256, 0, stream>>>(qb, kb);
    dim3 ga(LL / 16, HH, BB);
    flash_attn<<<ga, 32, 0, stream>>>(qb, kb, vb, maskA, ob);
    gemm_bf16<false,true,false,false><<<gD, 128, 0, stream>>>(ob, wo_i, nullptr, xc, xc, M, DD, DD);
    layernorm_bf16<<<M, 256, 0, stream>>>(xc, (size_t)DD, ln2g + (size_t)i * DD,
                                          ln2b + (size_t)i * DD, xnb, DD);
    gemm_bf16<true,false,true,true><<<gF, 128, 0, stream>>>(xnb, w1_i, b1 + (size_t)i * DFF, nullptr, hmid, M, DFF, DD);
    gemm_bf16<true,true,false,false><<<gD, 128, 0, stream>>>(hmid, w2_i, b2 + (size_t)i * DD, xc, xc, M, DD, DFF);
  }

  // Final LN + projection only for the 2 pooled rows (token 0 of each batch).
  layernorm_bf16<<<BB, 256, 0, stream>>>(xc, (size_t)LL * DD, outg, outb, pooled, DD);
  gemm_bf16<true,false,false,false><<<gp, 128, 0, stream>>>(pooled, pwb, projb, nullptr,
                                                            (float*)d_out, BB, DD, DD);
}
